// DualBranchFusionModel_14439680049643
// MI455X (gfx1250) — matrix operand order repaired and checked by reading, not compiled, not run on hardware
//
#include <hip/hip_runtime.h>

// ---------------------------------------------------------------------------
// Types for CDNA5 WMMA (wave32): V_WMMA_F32_16X16X32_BF16
// ---------------------------------------------------------------------------
typedef __attribute__((ext_vector_type(16))) __bf16 v16bf;
typedef __attribute__((ext_vector_type(8)))  float  v8f;

// hardware f32 -> bf16 (RTNE) via native convert
__device__ __forceinline__ unsigned short f2bf(float f) {
    union { __bf16 h; unsigned short s; } c; c.h = (__bf16)f; return c.s;
}

union AFrag { v16bf v; __bf16 h[16]; unsigned short s[16]; uint4 u[2]; };

// A-fragment per ISA 7.12.2 (16-bit A 16x32): lane halves hold K = kb..kb+7 and kb+16..kb+23
__device__ __forceinline__ void load_a(const unsigned short* row, int kb, AFrag& a) {
    a.u[0] = *(const uint4*)(row + kb);
    a.u[1] = *(const uint4*)(row + kb + 16);
}
__device__ __forceinline__ void load_a(const float* row, int kb, AFrag& a) {
#pragma unroll
    for (int i = 0; i < 8; ++i) a.h[i]     = (__bf16)row[kb + i];
#pragma unroll
    for (int i = 0; i < 8; ++i) a.h[8 + i] = (__bf16)row[kb + 16 + i];
}

// ---------------------------------------------------------------------------
// Weight pre-pack: W[K,NOUT] f32 -> fragment-ready bf16
// layout: out[((nt*KC + kc)*32 + lane)*16 + e], with n = nt*16 + (lane&15),
//         k = kc*32 + (lane>>4)*8 + (e&7) + (e>>3)*16   (B 32x16: matches A fragment K order)
// ---------------------------------------------------------------------------
template<int K, int NOUT>
__global__ void k_pack_b(const float* __restrict__ W, unsigned short* __restrict__ out) {
    constexpr int KC = K / 32;
    int idx = blockIdx.x * blockDim.x + threadIdx.x;
    if (idx >= K * NOUT) return;
    int e    = idx & 15;
    int lane = (idx >> 4) & 31;
    int rest = idx >> 9;
    int kc   = rest % KC;
    int nt   = rest / KC;
    int n = nt * 16 + (lane & 15);
    int k = kc * 32 + (lane >> 4) * 8 + (e & 7) + (e >> 3) * 16;
    out[idx] = f2bf(W[(size_t)k * NOUT + n]);
}

// ---------------------------------------------------------------------------
// Fused GEMM + bias + eval-BatchNorm + ReLU, WMMA bf16 -> f32 accum.
// One wave computes a 16-row x 64-col tile (4 N-tiles reuse one A fragment).
// A row index is clamped (not zero-filled): a garbage A row only affects the
// same output row, which is never stored -> no divergent EXEC around WMMA.
// ---------------------------------------------------------------------------
template<int K, int NOUT, typename AT, bool OUT_BF16>
__launch_bounds__(256)
__global__ void k_gin_mlp(const AT* __restrict__ A, const unsigned short* __restrict__ Bp,
                          const float* __restrict__ bias,
                          const float* __restrict__ bnb, const float* __restrict__ bng,
                          const float* __restrict__ bnm, const float* __restrict__ bnv,
                          void* __restrict__ outv, int nrows) {
    constexpr int KC  = K / 32;
    constexpr int NT  = NOUT / 16;
    constexpr int NTW = 4;                  // N-tiles per wave
    constexpr int NG  = NT / NTW;
    int mtiles = (nrows + 15) >> 4;
    int wid  = (int)((blockIdx.x * blockDim.x + threadIdx.x) >> 5);
    int lane = threadIdx.x & 31;
    if (wid >= mtiles * NG) return;         // uniform per wave -> EXEC stays full
    int mt = wid / NG, ng = wid % NG;
    int khalf  = lane >> 4;
    int arow_i = mt * 16 + (lane & 15);
    int ar     = arow_i < nrows ? arow_i : nrows - 1;   // clamp
    const AT* arow = A + (size_t)ar * K;

    v8f acc[NTW] = {};
#pragma unroll
    for (int kc = 0; kc < KC; ++kc) {
        AFrag a;
        load_a(arow, kc * 32 + khalf * 8, a);
#pragma unroll
        for (int t = 0; t < NTW; ++t) {
            int nt = ng * NTW + t;
            AFrag b;
            const uint4* bp = (const uint4*)(Bp + (((size_t)nt * KC + kc) * 32 + lane) * 16);
            b.u[0] = bp[0]; b.u[1] = bp[1];
            acc[t] = __builtin_amdgcn_wmma_f32_16x16x32_bf16(false, a.v, false, b.v,
                                                             (short)0, acc[t], false, false);
        }
    }

#pragma unroll
    for (int t = 0; t < NTW; ++t) {
        int n = (ng * NTW + t) * 16 + (lane & 15);
        float bi = bias[n];
        float mu = bnm[n], iv = rsqrtf(bnv[n] + 1e-5f), g = bng[n], be = bnb[n];
#pragma unroll
        for (int v = 0; v < 8; ++v) {
            int m = mt * 16 + khalf * 8 + v;   // D layout: lanes 0-15 rows 0-7, 16-31 rows 8-15
            if (m < nrows) {
                float y = fmaxf((acc[t][v] + bi - mu) * iv * g + be, 0.0f);
                if constexpr (OUT_BF16)
                    ((unsigned short*)outv)[(size_t)m * NOUT + n] = f2bf(y);
                else
                    ((float*)outv)[(size_t)m * NOUT + n] = y;
            }
        }
    }
}

// ---------------------------------------------------------------------------
// Elementwise / graph kernels
// ---------------------------------------------------------------------------
__global__ void k_embed(const float* __restrict__ xin, const float* __restrict__ W,
                        const float* __restrict__ bvec, float* __restrict__ xout, int nrows) {
    long long idx = (long long)blockIdx.x * blockDim.x + threadIdx.x;
    if (idx >= (long long)nrows * 128) return;
    int n = (int)(idx >> 7), c = (int)(idx & 127);
    const float* r = xin + (size_t)n * 9;
    float acc = bvec[c];
#pragma unroll
    for (int k = 0; k < 9; ++k) acc += r[k] * W[k * 128 + c];
    xout[idx] = acc;
}

__global__ void k_init_agg(float* __restrict__ agg, const float* __restrict__ x,
                           const float* __restrict__ eps, long long n) {
    long long idx = (long long)blockIdx.x * blockDim.x + threadIdx.x;
    if (idx < n) agg[idx] = (1.0f + eps[0]) * x[idx];
}

__global__ void k_scatter(float* __restrict__ agg, const float* __restrict__ x,
                          const int* __restrict__ src, const int* __restrict__ dst, int ne) {
    long long idx = (long long)blockIdx.x * blockDim.x + threadIdx.x;
    if (idx >= (long long)ne * 128) return;
    int e = (int)(idx >> 7), f = (int)(idx & 127);
    float v = x[(size_t)src[e] * 128 + f];
    __hip_atomic_fetch_add(&agg[(size_t)dst[e] * 128 + f], v,
                           __ATOMIC_RELAXED, __HIP_MEMORY_SCOPE_AGENT);
}

__global__ void k_zero(float* __restrict__ p, long long n) {
    long long idx = (long long)blockIdx.x * blockDim.x + threadIdx.x;
    if (idx < n) p[idx] = 0.0f;
}

__global__ void k_pool(const float* __restrict__ x, const int* __restrict__ batchv,
                       float* __restrict__ pooled, int nrows) {
    long long idx = (long long)blockIdx.x * blockDim.x + threadIdx.x;
    if (idx >= (long long)nrows * 128) return;
    int n = (int)(idx >> 7), f = (int)(idx & 127);
    __hip_atomic_fetch_add(&pooled[(size_t)batchv[n] * 128 + f], x[idx],
                           __ATOMIC_RELAXED, __HIP_MEMORY_SCOPE_AGENT);
}

__global__ void k_count(const int* __restrict__ batchv, float* __restrict__ cnt, int nrows) {
    int n = blockIdx.x * blockDim.x + threadIdx.x;
    if (n < nrows)
        __hip_atomic_fetch_add(&cnt[batchv[n]], 1.0f,
                               __ATOMIC_RELAXED, __HIP_MEMORY_SCOPE_AGENT);
}

__global__ void k_pooldiv(float* __restrict__ pooled, const float* __restrict__ cnt, long long n) {
    long long idx = (long long)blockIdx.x * blockDim.x + threadIdx.x;
    if (idx >= n) return;
    pooled[idx] /= fmaxf(cnt[idx >> 7], 1.0f);
}

// ---------------------------------------------------------------------------
// Head kernels (f32 VALU; tiny vs the GIN GEMMs)
// ---------------------------------------------------------------------------
__global__ void k_inter(const float* __restrict__ bs, const float* __restrict__ sv,
                        const float* __restrict__ W, const float* __restrict__ bvec,
                        const float* __restrict__ lnb, const float* __restrict__ lng,
                        float* __restrict__ out) {
    int b = blockIdx.x, c = threadIdx.x;              // 128 threads
    __shared__ float fin[256]; __shared__ float red[128]; __shared__ float mu_s, var_s;
    fin[c] = bs[(size_t)b * 128 + c];
    fin[128 + c] = sv[(size_t)b * 128 + c];
    __syncthreads();
    float acc = bvec[c];
    for (int j = 0; j < 256; ++j) acc += fin[j] * W[j * 128 + c];
    red[c] = acc; __syncthreads();
    for (int s = 64; s > 0; s >>= 1) { if (c < s) red[c] += red[c + s]; __syncthreads(); }
    if (c == 0) mu_s = red[0] / 128.0f; __syncthreads();
    float d = acc - mu_s; red[c] = d * d; __syncthreads();
    for (int s = 64; s > 0; s >>= 1) { if (c < s) red[c] += red[c + s]; __syncthreads(); }
    if (c == 0) var_s = red[0] / 128.0f; __syncthreads();
    out[(size_t)b * 128 + c] = fmaxf(d * rsqrtf(var_s + 1e-5f) * lng[c] + lnb[c], 0.0f);
}

__global__ void k_temp(const float* __restrict__ T, const float* __restrict__ w1,
                       const float* __restrict__ b1, const float* __restrict__ w2,
                       const float* __restrict__ b2, float* __restrict__ out) {
    int b = blockIdx.x, c = threadIdx.x;              // 32 threads
    __shared__ float t1[32];
    t1[c] = fmaxf(T[b] * w1[c] + b1[c], 0.0f);
    __syncthreads();
    float acc = b2[c];
    for (int j = 0; j < 32; ++j) acc += t1[j] * w2[j * 32 + c];
    out[(size_t)b * 32 + c] = acc;
}

__global__ void k_fp1(const float* __restrict__ aq, const float* __restrict__ inter,
                      const float* __restrict__ tf, const float* __restrict__ W,
                      const float* __restrict__ bvec, const float* __restrict__ lnb,
                      const float* __restrict__ lng, float* __restrict__ out) {
    int b = blockIdx.x, c = threadIdx.x;              // 128 threads
    __shared__ float fin[288]; __shared__ float red[128]; __shared__ float mu_s, var_s;
    fin[c] = aq[(size_t)b * 128 + c];
    fin[128 + c] = inter[(size_t)b * 128 + c];
    if (c < 32) fin[256 + c] = tf[(size_t)b * 32 + c];
    __syncthreads();
    float acc = bvec[c];
    for (int j = 0; j < 288; ++j) acc += fin[j] * W[j * 128 + c];
    red[c] = acc; __syncthreads();
    for (int s = 64; s > 0; s >>= 1) { if (c < s) red[c] += red[c + s]; __syncthreads(); }
    if (c == 0) mu_s = red[0] / 128.0f; __syncthreads();
    float d = acc - mu_s; red[c] = d * d; __syncthreads();
    for (int s = 64; s > 0; s >>= 1) { if (c < s) red[c] += red[c + s]; __syncthreads(); }
    if (c == 0) var_s = red[0] / 128.0f; __syncthreads();
    out[(size_t)b * 128 + c] = fmaxf(d * rsqrtf(var_s + 1e-5f) * lng[c] + lnb[c], 0.0f);
}

__global__ void k_fp2(const float* __restrict__ h1, const float* __restrict__ W,
                      const float* __restrict__ bvec, const float* __restrict__ lnb,
                      const float* __restrict__ lng, float* __restrict__ out) {
    int b = blockIdx.x, c = threadIdx.x;              // 64 threads
    __shared__ float fin[128]; __shared__ float red[64]; __shared__ float mu_s, var_s;
    fin[c] = h1[(size_t)b * 128 + c];
    fin[64 + c] = h1[(size_t)b * 128 + 64 + c];
    __syncthreads();
    float acc = bvec[c];
    for (int j = 0; j < 128; ++j) acc += fin[j] * W[j * 64 + c];
    red[c] = acc; __syncthreads();
    for (int s = 32; s > 0; s >>= 1) { if (c < s) red[c] += red[c + s]; __syncthreads(); }
    if (c == 0) mu_s = red[0] / 64.0f; __syncthreads();
    float d = acc - mu_s; red[c] = d * d; __syncthreads();
    for (int s = 32; s > 0; s >>= 1) { if (c < s) red[c] += red[c + s]; __syncthreads(); }
    if (c == 0) var_s = red[0] / 64.0f; __syncthreads();
    out[(size_t)b * 64 + c] = fmaxf(d * rsqrtf(var_s + 1e-5f) * lng[c] + lnb[c], 0.0f);
}

__global__ void k_fp3(const float* __restrict__ h2, const float* __restrict__ W,
                      const float* __restrict__ bvec, float* __restrict__ out) {
    int b = blockIdx.x, c = threadIdx.x;              // 64 threads
    __shared__ float red[64];
    red[c] = h2[(size_t)b * 64 + c] * W[c];
    __syncthreads();
    for (int s = 32; s > 0; s >>= 1) { if (c < s) red[c] += red[c + s]; __syncthreads(); }
    if (c == 0) out[b] = red[0] + bvec[0];
}

// ---------------------------------------------------------------------------
// Host orchestration
// ---------------------------------------------------------------------------
static inline unsigned cdiv(long long a, long long b) { return (unsigned)((a + b - 1) / b); }

// Encoder leaf order (JAX pytree, sorted dict keys):
//   0: embed.b  1: embed.w  then per layer l (base 2 + 13*l):
//   +0 bn_mid.beta +1 bn_mid.gamma +2 bn_mid.mean +3 bn_mid.var
//   +4 bn_out.beta +5 bn_out.gamma +6 bn_out.mean +7 bn_out.var
//   +8 eps  +9 lin1.b +10 lin1.w +11 lin2.b +12 lin2.w
static void run_encoder(const float* xin, const int* ei, const int* batchv, int Nn, int Ne,
                        void* const* d_in, int pbase,
                        float* xbuf, float* agg, unsigned short* midb,
                        unsigned short* wp1, unsigned short* wp2,
                        float* pooled, float* cnt, hipStream_t stream) {
    auto L = [&](int i) { return (const float*)d_in[pbase + i]; };

    k_embed<<<cdiv((long long)Nn * 128, 256), 256, 0, stream>>>(xin, L(1), L(0), xbuf, Nn);

    for (int l = 0; l < 4; ++l) {
        int lb = pbase + 2 + l * 13;
        auto LL = [&](int i) { return (const float*)d_in[lb + i]; };

        k_init_agg<<<cdiv((long long)Nn * 128, 256), 256, 0, stream>>>(
            agg, xbuf, LL(8), (long long)Nn * 128);
        k_scatter<<<cdiv((long long)Ne * 128, 256), 256, 0, stream>>>(
            agg, xbuf, ei, ei + Ne, Ne);

        k_pack_b<128, 256><<<cdiv(128 * 256, 256), 256, 0, stream>>>(LL(10), wp1);
        {
            int tiles = ((Nn + 15) / 16) * (256 / 16 / 4);   // 4 N-tiles per wave
            k_gin_mlp<128, 256, float, true><<<cdiv(tiles, 8), 256, 0, stream>>>(
                agg, wp1, LL(9), LL(0), LL(1), LL(2), LL(3), midb, Nn);
        }
        k_pack_b<256, 128><<<cdiv(256 * 128, 256), 256, 0, stream>>>(LL(12), wp2);
        {
            int tiles = ((Nn + 15) / 16) * (128 / 16 / 4);
            k_gin_mlp<256, 128, unsigned short, false><<<cdiv(tiles, 8), 256, 0, stream>>>(
                midb, wp2, LL(11), LL(4), LL(5), LL(6), LL(7), xbuf, Nn);
        }
    }

    k_zero<<<cdiv(4096 * 128, 256), 256, 0, stream>>>(pooled, 4096LL * 128);
    k_zero<<<cdiv(4096, 256), 256, 0, stream>>>(cnt, 4096);
    k_pool<<<cdiv((long long)Nn * 128, 256), 256, 0, stream>>>(xbuf, batchv, pooled, Nn);
    k_count<<<cdiv(Nn, 256), 256, 0, stream>>>(batchv, cnt, Nn);
    k_pooldiv<<<cdiv(4096 * 128, 256), 256, 0, stream>>>(pooled, cnt, 4096LL * 128);
}

extern "C" void kernel_launch(void* const* d_in, const int* in_sizes, int n_in,
                              void* d_out, int out_size, void* d_ws, size_t ws_size,
                              hipStream_t stream) {
    (void)n_in; (void)out_size; (void)ws_size;
    const float* drug_x   = (const float*)d_in[0];
    const int*   drug_ei  = (const int*)d_in[1];
    const int*   drug_b   = (const int*)d_in[2];
    const float* solv_x   = (const float*)d_in[3];
    const int*   solv_ei  = (const int*)d_in[4];
    const int*   solv_b   = (const int*)d_in[5];
    const float* temper   = (const float*)d_in[6];
    const int N_DRUG = in_sizes[0] / 9;
    const int E_DRUG = in_sizes[1] / 2;
    const int N_SOLV = in_sizes[3] / 9;
    const int E_SOLV = in_sizes[4] / 2;
    const int Bsz    = in_sizes[6];
    const int P = 7;  // params leaves start; sorted keys: enc_aq, enc_bs, enc_solv, fp_*, inter_*, temp*

    // workspace carve (256B aligned)
    char* w = (char*)d_ws;
    auto alloc = [&](size_t bytes) -> void* {
        void* p = (void*)w; w += (bytes + 255) & ~(size_t)255; return p;
    };
    int NMAX = N_DRUG > N_SOLV ? N_DRUG : N_SOLV;
    float*          xbuf   = (float*)alloc((size_t)NMAX * 128 * 4);
    float*          agg    = (float*)alloc((size_t)NMAX * 128 * 4);
    unsigned short* midb   = (unsigned short*)alloc((size_t)NMAX * 256 * 2);
    unsigned short* wp1    = (unsigned short*)alloc(128 * 256 * 2);
    unsigned short* wp2    = (unsigned short*)alloc(256 * 128 * 2);
    float*          p_aq   = (float*)alloc((size_t)Bsz * 128 * 4);
    float*          p_bs   = (float*)alloc((size_t)Bsz * 128 * 4);
    float*          p_sv   = (float*)alloc((size_t)Bsz * 128 * 4);
    float*          cnt    = (float*)alloc((size_t)Bsz * 4);
    float*          interb = (float*)alloc((size_t)Bsz * 128 * 4);
    float*          tfeat  = (float*)alloc((size_t)Bsz * 32 * 4);
    float*          h1     = (float*)alloc((size_t)Bsz * 128 * 4);
    float*          h2     = (float*)alloc((size_t)Bsz * 64 * 4);

    // three GIN encoders (54 leaves each)
    run_encoder(drug_x, drug_ei, drug_b, N_DRUG, E_DRUG, d_in, P +   0,
                xbuf, agg, midb, wp1, wp2, p_aq, cnt, stream);
    run_encoder(drug_x, drug_ei, drug_b, N_DRUG, E_DRUG, d_in, P +  54,
                xbuf, agg, midb, wp1, wp2, p_bs, cnt, stream);
    run_encoder(solv_x, solv_ei, solv_b, N_SOLV, E_SOLV, d_in, P + 108,
                xbuf, agg, midb, wp1, wp2, p_sv, cnt, stream);

    // head leaf indices (sorted keys after encoders):
    // fp_lin1: b=162 w=163 | fp_lin2: b=164 w=165 | fp_lin3: b=166 w=167
    // fp_ln1: beta=168 gamma=169 | fp_ln2: beta=170 gamma=171
    // inter_lin: b=172 w=173 | inter_ln: beta=174 gamma=175
    // temp1: b=176 w=177 | temp2: b=178 w=179
    auto G = [&](int i) { return (const float*)d_in[P + i]; };

    k_inter<<<Bsz, 128, 0, stream>>>(p_bs, p_sv, G(173), G(172), G(174), G(175), interb);
    k_temp<<<Bsz, 32, 0, stream>>>(temper, G(177), G(176), G(179), G(178), tfeat);
    k_fp1<<<Bsz, 128, 0, stream>>>(p_aq, interb, tfeat, G(163), G(162), G(168), G(169), h1);
    k_fp2<<<Bsz, 64, 0, stream>>>(h1, G(165), G(164), G(170), G(171), h2);
    k_fp3<<<Bsz, 64, 0, stream>>>(h2, G(167), G(166), (float*)d_out);
}
